// DCRNNModel_57062935495050
// MI455X (gfx1250) — compile-verified, matrix operand order checked by reading, not run on hardware
//
#include <hip/hip_runtime.h>
#include <hip/hip_bf16.h>
#include <math.h>

// Problem constants (match reference)
#define NN    512   // nodes
#define BB    32    // batch
#define TT    12    // encoder steps
#define HORZ  12    // decoder horizon
#define UU    64    // rnn units
#define MM    5     // num diffusion matrices (1 + 2 supports * K)
#define CPMAX 128   // padded channel stride (layer1 C=128; layer0 C=65 -> Cp=96, stride 128)

typedef __bf16 bf16_t;
typedef bf16_t bf16x8 __attribute__((ext_vector_type(8)));
typedef bf16_t v16bf  __attribute__((ext_vector_type(16)));
typedef float  v8f    __attribute__((ext_vector_type(8)));
typedef int    i32x4  __attribute__((ext_vector_type(4)));
typedef int    i32x8  __attribute__((ext_vector_type(8)));

union FragU { v16bf v; bf16x8 h[2]; };

// Load a 16x32 bf16 WMMA operand fragment (A layout; B uses the same per-lane
// pattern with "row" = output column). Element (row, k) at base[row*ld + k].
// lane<16: row=lane,    K chunks [0..7]  and [16..23]
// lane>=16: row=lane-16, K chunks [8..15] and [24..31]
__device__ __forceinline__ v16bf load_frag(const bf16_t* base, int ld, int lane) {
  const int r  = lane & 15;
  const int ko = (lane >> 4) << 3;          // 0 or 8
  const bf16_t* p = base + (size_t)r * ld + ko;
  FragU f;
  f.h[0] = *(const bf16x8*)(p);             // K = ko .. ko+7      (16B)
  f.h[1] = *(const bf16x8*)(p + 16);        // K = ko+16 .. ko+23  (16B)
  return f.v;
}

// ---------------------------------------------------------------------------
// TDM: async 2D tile load Global -> LDS via Tensor Data Mover (gfx1250).
// D# per CDNA5 ISA:
//   group0: [1:0]=count=1, [63:32]=lds_addr, [120:64]=global_addr, [127:126]=2
//   group1: [17:16]=data_size(1 -> 2B), [79:48]=tensor_dim0, [111:80]=tensor_dim1,
//           [127:112]=tile_dim0, [143:128]=tile_dim1, [207:160]=tensor_dim0_stride
// LDS byte offset = low 32 bits of the generic pointer (ISA §10.2: LDS flat
// addresses truncate to addr[31:0]; the LDS aperture base has zero low bits).
// ---------------------------------------------------------------------------
__device__ __forceinline__ void tdm_load_2d(const void* gsrc, uint32_t lds_off,
                                            int dim0, int rows, int stride_elems) {
  uint64_t ga = (uint64_t)(uintptr_t)gsrc;
  i32x4 g0;
  g0[0] = 1;                                           // count=1 (valid descriptor)
  g0[1] = (int)lds_off;                                // lds_addr
  g0[2] = (int)(uint32_t)ga;                           // global_addr[31:0]
  g0[3] = (int)(((uint32_t)(ga >> 32) & 0x01FFFFFFu)   // global_addr[56:32]
                | 0x80000000u);                        // type=2 ("image")
  i32x8 g1;
  g1[0] = 0x00010000;                                  // data_size=1 (2 bytes/elem)
  g1[1] = (dim0 & 0xFFFF) << 16;                       // tensor_dim0[15:0]
  g1[2] = ((dim0 >> 16) & 0xFFFF) | ((rows & 0xFFFF) << 16); // dim0 hi | tensor_dim1 lo
  g1[3] = ((rows >> 16) & 0xFFFF) | ((dim0 & 0xFFFF) << 16); // dim1 hi | tile_dim0
  g1[4] = rows & 0xFFFF;                               // tile_dim1 (tile_dim2 = 0)
  g1[5] = stride_elems;                                // tensor_dim0_stride[31:0]
  g1[6] = 0;
  g1[7] = 0;                                           // tensor_dim1_stride unused (2D)
  asm volatile("tensor_load_to_lds %0, %1" :: "s"(g0), "s"(g1) : "memory");
}

// ---------------------------------------------------------------------------
// Diffusion GEMM: one block per (b, n-tile). TDM stages the shared 16x512
// strip of S into LDS once; 4 waves loop over c-tiles, A-frags from LDS
// (ds_load_b128), B-frags from global (b128). K = 512 in 16 WMMA steps.
// If X0cn != null computes 2*acc - x0 (Chebyshev). Writes CN copy (b128)
// and NC transpose copy (b16 strided); zero-fills NC K-padding.
// ---------------------------------------------------------------------------
__global__ void k_diffusion(const bf16_t* __restrict__ S,
                            const bf16_t* __restrict__ Bcn,     // [B][CPMAX][NN]
                            const bf16_t* __restrict__ X0cn,    // optional, same layout
                            bf16_t* __restrict__ OutCN,         // optional, same layout
                            bf16_t* __restrict__ XncSlice,      // [B][NN][CPMAX]
                            int Ctiles, int Ctot)
{
  __shared__ __align__(16) bf16_t shA[16 * NN];        // 16KB: S rows n0..n0+15
  const int lane = threadIdx.x & 31;
  const int warp = threadIdx.x >> 5;
  const int nt = blockIdx.x & 31;
  const int b  = blockIdx.x >> 5;
  const int n0 = nt << 4;

  if (warp == 0) {
    tdm_load_2d(S + (size_t)n0 * NN, (uint32_t)(uintptr_t)shA, NN, 16, NN);
    __builtin_amdgcn_s_wait_tensorcnt(0);
  }
  __syncthreads();

  const bf16_t* Bb = Bcn + (size_t)b * CPMAX * NN;

  for (int ct = warp; ct < Ctiles; ct += 4) {
    const int c0 = ct << 4;
    v8f acc = {};
#pragma unroll 4
    for (int k = 0; k < NN; k += 32) {
      v16bf a  = load_frag(shA + k, NN, lane);                  // LDS (ds_load_b128)
      v16bf bb = load_frag(Bb + (size_t)c0 * NN + k, NN, lane); // global b128
      acc = __builtin_amdgcn_wmma_f32_16x16x32_bf16(false, a, false, bb,
                                                    (short)0, acc, false, false);
    }

    // D layout: lane holds col c = c0+(lane&15), rows n = n0 + 8*(lane>>4) + r
    const int c  = c0 + (lane & 15);
    const int nb = n0 + ((lane >> 4) << 3);

    float res[8];
    if (X0cn != nullptr) {
      bf16x8 x0 = *(const bf16x8*)(X0cn + ((size_t)b * CPMAX + c) * NN + nb);
#pragma unroll
      for (int r = 0; r < 8; ++r) res[r] = 2.0f * acc[r] - (float)x0[r];
    } else {
#pragma unroll
      for (int r = 0; r < 8; ++r) res[r] = acc[r];
    }

    bf16_t* xn = XncSlice + ((size_t)b * NN + nb) * CPMAX + c;
    if (c < Ctot) {
      bf16x8 o;
#pragma unroll
      for (int r = 0; r < 8; ++r) o[r] = (bf16_t)res[r];
      if (OutCN)
        *(bf16x8*)(OutCN + ((size_t)b * CPMAX + c) * NN + nb) = o; // contiguous in n
#pragma unroll
      for (int r = 0; r < 8; ++r) xn[(size_t)r * CPMAX] = o[r];    // NC transpose copy
    } else {
#pragma unroll
      for (int r = 0; r < 8; ++r) xn[(size_t)r * CPMAX] = (bf16_t)0.0f; // zero K-pad
    }
  }
}

// ---------------------------------------------------------------------------
// Weight GEMM (Out^T tiles): Out[b][n][o] = act( sum_m Wt[m] . X_m + bias ).
// One block per (b, n-tile). TDM stages the 5 shared 16xCPMAX activation
// strips into LDS once; 4 waves loop over o-tiles, A-frags (weights) from
// global, B-frags from LDS. D rows = o, cols = n -> one b128 NC store/lane.
// ---------------------------------------------------------------------------
__global__ void k_wgemm(const bf16_t* __restrict__ Wt,   // [MM][O][CPMAX]
                        const bf16_t* __restrict__ Xnc,  // [MM][BB][NN][CPMAX]
                        const float*  __restrict__ bias, // [O]
                        bf16_t* __restrict__ OutNC,      // [BB][NN][O]
                        int O, int Cp, int act)          // act: 0=sigmoid 1=tanh
{
  __shared__ __align__(16) bf16_t shB[MM * 16 * CPMAX]; // 20KB
  const int lane = threadIdx.x & 31;
  const int warp = threadIdx.x >> 5;
  const int nt = blockIdx.x & 31;
  const int b  = blockIdx.x >> 5;
  const int n0 = nt << 4;

  if (warp == 0) {
    for (int m = 0; m < MM; ++m) {
      const bf16_t* src = Xnc + ((size_t)m * BB + b) * NN * CPMAX + (size_t)n0 * CPMAX;
      tdm_load_2d(src, (uint32_t)(uintptr_t)shB + (uint32_t)(m * 16 * CPMAX * 2),
                  CPMAX, 16, CPMAX);
    }
    __builtin_amdgcn_s_wait_tensorcnt(0);
  }
  __syncthreads();

  const int Ot = O >> 4;
  for (int ot = warp; ot < Ot; ot += 4) {
    const int o0 = ot << 4;
    v8f acc = {};
    for (int m = 0; m < MM; ++m) {
      const bf16_t* Am = Wt + ((size_t)m * O + o0) * CPMAX;
      const bf16_t* Bm = shB + m * 16 * CPMAX;
#pragma unroll 4
      for (int k = 0; k < Cp; k += 32) {
        v16bf a  = load_frag(Am + k, CPMAX, lane);   // global b128
        v16bf bb = load_frag(Bm + k, CPMAX, lane);   // LDS ds_load_b128
        acc = __builtin_amdgcn_wmma_f32_16x16x32_bf16(false, a, false, bb,
                                                      (short)0, acc, false, false);
      }
    }

    const int n  = n0 + (lane & 15);
    const int ob = o0 + ((lane >> 4) << 3);
    bf16x8 out;
#pragma unroll
    for (int r = 0; r < 8; ++r) {
      float v = acc[r] + bias[ob + r];
      v = act ? tanhf(v) : (1.0f / (1.0f + __expf(-v)));
      out[r] = (bf16_t)v;
    }
    *(bf16x8*)(OutNC + ((size_t)b * NN + n) * O + ob) = out;
  }
}

// ---------------------------------------------------------------------------
// Elementwise / prep kernels
// ---------------------------------------------------------------------------
// Build x0 = concat(x, h) in both CN and NC layouts, with zero K-padding.
// mode 0: x = f32 [B][NN] (encoder input slice), Cx==1
// mode 1: x = bf16 [B][NN] (decoder feedback),   Cx==1
// mode 2: x = bf16 CN [B][UU][NN] (h0 -> layer1), Cx==UU
__global__ void k_concat(const void* __restrict__ xsrc, int mode,
                         const bf16_t* __restrict__ Hcn,
                         bf16_t* __restrict__ Xcn0, bf16_t* __restrict__ Xnc0,
                         int Cx, int Ctot, int Cp)
{
  size_t i = (size_t)blockIdx.x * blockDim.x + threadIdx.x;
  size_t total = (size_t)BB * Cp * NN;
  if (i >= total) return;
  int n = (int)(i % NN);
  int c = (int)((i / NN) % Cp);
  int b = (int)(i / ((size_t)NN * Cp));
  float v = 0.0f;
  if (c < Cx) {
    if (mode == 0)      v = ((const float*)xsrc)[(size_t)b * NN + n];
    else if (mode == 1) v = (float)((const bf16_t*)xsrc)[(size_t)b * NN + n];
    else                v = (float)((const bf16_t*)xsrc)[((size_t)b * UU + c) * NN + n];
  } else if (c < Ctot) {
    v = (float)Hcn[((size_t)b * UU + (c - Cx)) * NN + n];
  }
  bf16_t bv = (bf16_t)v;
  Xcn0[((size_t)b * CPMAX + c) * NN + n] = bv;
  Xnc0[((size_t)b * NN + n) * CPMAX + c] = bv;
}

__global__ void k_rh(const bf16_t* __restrict__ G,    // gate NC [B][NN][2U], sigmoid applied
                     const bf16_t* __restrict__ Hcn,  // [B][UU][NN]
                     bf16_t* __restrict__ RHcn)       // [B][UU][NN]
{
  size_t i = (size_t)blockIdx.x * blockDim.x + threadIdx.x;
  if (i >= (size_t)BB * UU * NN) return;
  int n = (int)(i % NN);
  int c = (int)((i / NN) % UU);
  int b = (int)(i / ((size_t)NN * UU));
  float r = (float)G[((size_t)b * NN + n) * (2 * UU) + c];
  float h = (float)Hcn[((size_t)b * UU + c) * NN + n];
  RHcn[((size_t)b * UU + c) * NN + n] = (bf16_t)(r * h);
}

__global__ void k_hupd(const bf16_t* __restrict__ G,   // gate NC [B][NN][2U]
                       const bf16_t* __restrict__ Gc,  // cand NC [B][NN][U], tanh applied
                       bf16_t* __restrict__ Hcn)       // [B][UU][NN] in/out
{
  size_t i = (size_t)blockIdx.x * blockDim.x + threadIdx.x;
  if (i >= (size_t)BB * UU * NN) return;
  int n = (int)(i % NN);
  int c = (int)((i / NN) % UU);
  int b = (int)(i / ((size_t)NN * UU));
  float u  = (float)G[((size_t)b * NN + n) * (2 * UU) + UU + c];
  float cd = (float)Gc[((size_t)b * NN + n) * UU + c];
  size_t hi = ((size_t)b * UU + c) * NN + n;
  float h = (float)Hcn[hi];
  Hcn[hi] = (bf16_t)(u * h + (1.0f - u) * cd);
}

__global__ void k_proj(const bf16_t* __restrict__ Hcn, const float* __restrict__ pw,
                       const float* __restrict__ pb, float* __restrict__ out,
                       bf16_t* __restrict__ xin)
{
  int i = blockIdx.x * blockDim.x + threadIdx.x;     // BB*NN
  if (i >= BB * NN) return;
  int n = i % NN, b = i / NN;
  float acc = pb[0];
#pragma unroll 8
  for (int c = 0; c < UU; ++c)
    acc += (float)Hcn[((size_t)b * UU + c) * NN + n] * pw[c];
  out[i] = acc;
  xin[i] = (bf16_t)acc;
}

__global__ void k_f32_to_bf16(const float* __restrict__ src, bf16_t* __restrict__ dst,
                              size_t nElem)
{
  size_t i = (size_t)blockIdx.x * blockDim.x + threadIdx.x;
  if (i < nElem) dst[i] = (bf16_t)src[i];
}

__global__ void k_zero_bf16(bf16_t* __restrict__ dst, size_t nElem)
{
  size_t i = (size_t)blockIdx.x * blockDim.x + threadIdx.x;
  if (i < nElem) dst[i] = (bf16_t)0.0f;
}

// Wt[m][o][c] = W[c*MM + m][o] for c < Ctot, else 0 (zero K-pad).
__global__ void k_prep_w(const float* __restrict__ W, bf16_t* __restrict__ Wt,
                         int Ctot, int O)
{
  size_t i = (size_t)blockIdx.x * blockDim.x + threadIdx.x;
  size_t total = (size_t)MM * O * CPMAX;
  if (i >= total) return;
  int c = (int)(i % CPMAX);
  int o = (int)((i / CPMAX) % O);
  int m = (int)(i / ((size_t)CPMAX * O));
  float v = (c < Ctot) ? W[((size_t)c * MM + m) * O + o] : 0.0f;
  Wt[i] = (bf16_t)v;
}

// ---------------------------------------------------------------------------
// Host orchestration
// ---------------------------------------------------------------------------
static inline size_t alignup(size_t x) { return (x + 255) & ~(size_t)255; }

extern "C" void kernel_launch(void* const* d_in, const int* in_sizes, int n_in,
                              void* d_out, int out_size, void* d_ws, size_t ws_size,
                              hipStream_t stream) {
  (void)in_sizes; (void)n_in; (void)out_size; (void)ws_size;

  const float* in_inputs = (const float*)d_in[0];
  const float* in_sup    = (const float*)d_in[1];
  const float* projW     = (const float*)d_in[18];
  const float* projB     = (const float*)d_in[19];

  // Workspace carve-up (~42 MB total; everything L2-resident)
  char* p = (char*)d_ws;
  auto take = [&](size_t bytes) -> void* { void* q = (void*)p; p += alignup(bytes); return q; };

  bf16_t* Sbf = (bf16_t*)take((size_t)2 * NN * NN * 2);
  bf16_t* WtG[4]; bf16_t* WtC[4];
  for (int cid = 0; cid < 4; ++cid) {
    WtG[cid] = (bf16_t*)take((size_t)MM * 128 * CPMAX * 2);
    WtC[cid] = (bf16_t*)take((size_t)MM * 64  * CPMAX * 2);
  }
  bf16_t* h0   = (bf16_t*)take((size_t)BB * UU * NN * 2);
  bf16_t* h1   = (bf16_t*)take((size_t)BB * UU * NN * 2);
  bf16_t* Xcn0 = (bf16_t*)take((size_t)BB * CPMAX * NN * 2);
  bf16_t* X1cn = (bf16_t*)take((size_t)BB * CPMAX * NN * 2);
  bf16_t* Xnc  = (bf16_t*)take((size_t)MM * BB * NN * CPMAX * 2);
  bf16_t* Gg   = (bf16_t*)take((size_t)BB * NN * 128 * 2);
  bf16_t* Gc   = (bf16_t*)take((size_t)BB * NN * 64 * 2);
  bf16_t* RH   = (bf16_t*)take((size_t)BB * UU * NN * 2);
  bf16_t* xin  = (bf16_t*)take((size_t)BB * NN * 2);

  // ---- prep: bf16 supports, transposed/sliced/zero-padded weights, zero h ----
  {
    size_t ns = (size_t)2 * NN * NN;
    k_f32_to_bf16<<<dim3((unsigned)((ns + 255) / 256)), 256, 0, stream>>>(in_sup, Sbf, ns);
    const int CtotL[4] = {65, 128, 65, 128};
    for (int cid = 0; cid < 4; ++cid) {
      size_t tg = (size_t)MM * 128 * CPMAX;
      size_t tc = (size_t)MM * 64 * CPMAX;
      k_prep_w<<<dim3((unsigned)((tg + 255) / 256)), 256, 0, stream>>>(
          (const float*)d_in[2 + 4 * cid], WtG[cid], CtotL[cid], 128);
      k_prep_w<<<dim3((unsigned)((tc + 255) / 256)), 256, 0, stream>>>(
          (const float*)d_in[4 + 4 * cid], WtC[cid], CtotL[cid], 64);
    }
    size_t nh = (size_t)BB * UU * NN;
    k_zero_bf16<<<dim3((unsigned)((nh + 255) / 256)), 256, 0, stream>>>(h0, nh);
    k_zero_bf16<<<dim3((unsigned)((nh + 255) / 256)), 256, 0, stream>>>(h1, nh);
  }

  // ---- one DCGRU cell ----
  auto run_cell = [&](int mode, const void* xsrc, bf16_t* H, int cid, int Ctot, int Cp) {
    const int Cx = Ctot - UU;
    const int Ctiles = Cp >> 4;
    const float* bg = (const float*)d_in[3 + 4 * cid];
    const float* bc = (const float*)d_in[5 + 4 * cid];

    const size_t ccount = (size_t)BB * Cp * NN;
    const dim3 cgrid((unsigned)((ccount + 255) / 256));
    const dim3 tgrid((unsigned)(BB * 32));               // one block per (b, n-tile)
    const size_t ecount = (size_t)BB * UU * NN;
    const dim3 egrid((unsigned)((ecount + 255) / 256));

    auto diffuse = [&]() {
      for (int s = 0; s < 2; ++s) {
        const bf16_t* Ss = Sbf + (size_t)s * NN * NN;
        bf16_t* slice1 = Xnc + (size_t)(1 + 2 * s) * BB * NN * CPMAX;
        bf16_t* slice2 = Xnc + (size_t)(2 + 2 * s) * BB * NN * CPMAX;
        k_diffusion<<<tgrid, 128, 0, stream>>>(Ss, Xcn0, nullptr, X1cn, slice1, Ctiles, Ctot);
        k_diffusion<<<tgrid, 128, 0, stream>>>(Ss, X1cn, Xcn0, nullptr, slice2, Ctiles, Ctot);
      }
    };

    // gate gconv: g = sigmoid(gconv(concat(x, h)))
    k_concat<<<cgrid, 256, 0, stream>>>(xsrc, mode, H, Xcn0, Xnc, Cx, Ctot, Cp);
    diffuse();
    k_wgemm<<<tgrid, 128, 0, stream>>>(WtG[cid], Xnc, bg, Gg, 128, Cp, /*sigmoid*/0);
    k_rh<<<egrid, 256, 0, stream>>>(Gg, H, RH);

    // candidate gconv: c = tanh(gconv(concat(x, r*h)))
    k_concat<<<cgrid, 256, 0, stream>>>(xsrc, mode, RH, Xcn0, Xnc, Cx, Ctot, Cp);
    diffuse();
    k_wgemm<<<tgrid, 128, 0, stream>>>(WtC[cid], Xnc, bc, Gc, 64, Cp, /*tanh*/1);

    // h = u*h + (1-u)*c
    k_hupd<<<egrid, 256, 0, stream>>>(Gg, Gc, H);
  };

  // ---- encoder ----
  for (int t = 0; t < TT; ++t) {
    run_cell(0, in_inputs + (size_t)t * BB * NN, h0, 0, 65, 96);
    run_cell(2, h0, h1, 1, 128, 128);
  }

  // ---- decoder (go symbol = zeros) ----
  {
    size_t nx = (size_t)BB * NN;
    k_zero_bf16<<<dim3((unsigned)((nx + 255) / 256)), 256, 0, stream>>>(xin, nx);
  }
  for (int t = 0; t < HORZ; ++t) {
    run_cell(1, xin, h0, 2, 65, 96);
    run_cell(2, h0, h1, 3, 128, 128);
    k_proj<<<dim3((BB * NN + 255) / 256), 256, 0, stream>>>(
        h1, projW, projB, (float*)d_out + (size_t)t * BB * NN, xin);
  }
}
